// SMI_10307921511138
// MI455X (gfx1250) — matrix operand order repaired and checked by reading, not compiled, not run on hardware
//
#include <hip/hip_runtime.h>

typedef __bf16 bf16_t;
typedef __attribute__((ext_vector_type(16))) __bf16 v16bf;
typedef __attribute__((ext_vector_type(8)))  float  v8f;

#define NEGV (-1000000000.0f)
#define INV_SQRT_DL 0.08838834764831845f   // 1/sqrt(128)
#define INV_SQRT_D  0.044194173824159216f  // 1/sqrt(512)

__device__ inline float wave_sum(float v) {
  #pragma unroll
  for (int off = 16; off > 0; off >>= 1) v += __shfl_xor(v, off, 32);
  return v;
}

enum { AM_DIRECT = 0, AM_OUTER = 1, AM_MEAN4 = 2 };
enum { EM_BIAS = 0, EM_MASK_BST = 1, EM_MASK_ROW = 2, EM_CC = 3, EM_MOM = 4 };

// ---------------------------------------------------------------------------
// Generic bf16 WMMA GEMM: C[M,N] = epilogue(A[M,Kd] @ Wt^T + bias)
//   Wt is bf16, stored transposed: Wt[n*Kd + k]
//   One wave -> 16x64 output strip (4 accumulators sharing one A fragment).
//   A-fragment layout (16-bit A 16x32): lanes 0-15 row M=l15, K = k0+0..15 ;
//   lanes 16-31 same rows, K = k0+16..31.
//   B-fragment: lane = column N, lanes 0-15 K=k0..k0+15, lanes 16-31 K=+16.
// ---------------------------------------------------------------------------
template<int AM, int EM>
__global__ __launch_bounds__(256)
void k_gemm(const float* __restrict__ A, const bf16_t* __restrict__ Wt,
            const float* __restrict__ bias, float* __restrict__ C,
            int M, int Kd, int N,
            const float* __restrict__ mask,
            const float* __restrict__ mu,
            const float* __restrict__ fs)
{
  const int lane = threadIdx.x & 31;
  const int l15  = lane & 15;
  const int half = lane >> 4;
  const int tilesNg = N >> 6;                 // groups of 4 n-tiles
  const int tiles   = (M >> 4) * tilesNg;
  int tid = (blockIdx.x << 3) + (threadIdx.x >> 5);
  if (tid >= tiles) return;                   // wave-uniform exit (EXEC all-1 for WMMA)
  int mt  = tid / tilesNg;
  int nt0 = (tid - mt * tilesNg) << 2;

  int am = (mt << 4) + l15;                   // A row owned by this lane
  const float* r1;
  const float* r2 = nullptr;
  if constexpr (AM == AM_DIRECT) {
    r1 = A + (size_t)am * Kd;
  } else if constexpr (AM == AM_OUTER) {      // A[m,k] = fb[b,i,k]*fb[b,j,k]
    int b = am >> 10, i = (am >> 5) & 31, j = am & 31;
    r1 = A + (size_t)(b * 32 + i) * Kd;
    r2 = A + (size_t)(b * 32 + j) * Kd;
  } else {                                    // A[m,k] = mean over 4 rows
    r1 = A + (size_t)am * 4 * Kd;
  }
  const int kshift = half << 4;

  const bf16_t* w0 = Wt + (size_t)((nt0 + 0) * 16 + l15) * Kd + (half << 4);
  const bf16_t* w1 = Wt + (size_t)((nt0 + 1) * 16 + l15) * Kd + (half << 4);
  const bf16_t* w2 = Wt + (size_t)((nt0 + 2) * 16 + l15) * Kd + (half << 4);
  const bf16_t* w3 = Wt + (size_t)((nt0 + 3) * 16 + l15) * Kd + (half << 4);

  v8f accs[4];
  #pragma unroll
  for (int q = 0; q < 4; q++) accs[q] = (v8f){0.f,0.f,0.f,0.f,0.f,0.f,0.f,0.f};

  for (int k0 = 0; k0 < Kd; k0 += 32) {
    v16bf af;
    const float* p = r1 + k0 + kshift;
    if constexpr (AM == AM_DIRECT) {
      #pragma unroll
      for (int i = 0; i < 8; i++) { af[i] = (bf16_t)p[i]; af[i+8] = (bf16_t)p[i+8]; }
    } else if constexpr (AM == AM_OUTER) {
      const float* p2 = r2 + k0 + kshift;
      #pragma unroll
      for (int i = 0; i < 8; i++) {
        af[i]   = (bf16_t)(p[i]    * p2[i]);
        af[i+8] = (bf16_t)(p[i+8] * p2[i+8]);
      }
    } else {
      #pragma unroll
      for (int i = 0; i < 8; i++) {
        af[i]   = (bf16_t)(0.25f * (p[i]    + p[i+Kd]    + p[i+2*Kd]    + p[i+3*Kd]));
        af[i+8] = (bf16_t)(0.25f * (p[i+8] + p[i+8+Kd] + p[i+8+2*Kd] + p[i+8+3*Kd]));
      }
    }
    v16bf b0 = *(const v16bf*)(w0 + k0);
    v16bf b1 = *(const v16bf*)(w1 + k0);
    v16bf b2 = *(const v16bf*)(w2 + k0);
    v16bf b3 = *(const v16bf*)(w3 + k0);
    accs[0] = __builtin_amdgcn_wmma_f32_16x16x32_bf16(false, af, false, b0, (short)0, accs[0], false, false);
    accs[1] = __builtin_amdgcn_wmma_f32_16x16x32_bf16(false, af, false, b1, (short)0, accs[1], false, false);
    accs[2] = __builtin_amdgcn_wmma_f32_16x16x32_bf16(false, af, false, b2, (short)0, accs[2], false, false);
    accs[3] = __builtin_amdgcn_wmma_f32_16x16x32_bf16(false, af, false, b3, (short)0, accs[3], false, false);
  }

  // C/D layout: VGPR r, lanes 0-15 -> M = r, lanes 16-31 -> M = r+8; N = l15.
  #pragma unroll
  for (int q = 0; q < 4; q++) {
    int nn = (nt0 + q) * 16 + l15;
    float bn = bias[nn];
    #pragma unroll
    for (int r = 0; r < 8; r++) {
      int mm = (mt << 4) + r + (half << 3);
      size_t idx = (size_t)mm * N + nn;
      float v = accs[q][r] + bn;
      if constexpr (EM == EM_BIAS) {
        C[idx] = v;
      } else if constexpr (EM == EM_MASK_BST) {        // f_c_hat: *mmask[bst]
        C[idx] = v * mask[mm >> 2];
      } else if constexpr (EM == EM_MASK_ROW) {        // f_w_hat: *qmask[row]
        C[idx] = v * mask[mm];
      } else if constexpr (EM == EM_CC) {              // cu = f_cc + cu + gm*mu
        int bst = mm >> 2; int b = mm >> 12;
        v *= mask[bst];
        float x = mu[(size_t)bst * N + nn];
        float s = fs[b * N + nn];
        float g = 1.f / (1.f + __expf(-x * s));
        C[idx] = v + C[idx] + g * x;
      } else {                                         // EM_MOM: mu += conv
        C[idx] += v * mask[mm];
      }
    }
  }
}

// ---------------------------------------------------------------------------
// Weight fp32 [K,N] -> bf16 transposed [N,K]
// ---------------------------------------------------------------------------
__global__ __launch_bounds__(256)
void k_w2t(const float* __restrict__ W, bf16_t* __restrict__ Wt, int K, int N)
{
  int t = blockIdx.x * 256 + threadIdx.x;
  if (t >= K * N) return;
  int k = t / N, n = t - k * N;
  Wt[(size_t)n * K + k] = (bf16_t)W[t];
}

// f_s_hat[b,n] = f_s[b,:] @ Wsh + bsh   (16x128 outputs)
__global__ __launch_bounds__(256)
void k_fsh(const float* __restrict__ fs, const float* __restrict__ Wsh,
           const float* __restrict__ bsh, float* __restrict__ fsh)
{
  int t = blockIdx.x * 256 + threadIdx.x;
  if (t >= 2048) return;
  int b = t >> 7, n = t & 127;
  float s = bsh[n];
  for (int k = 0; k < 512; k++) s += fs[b * 512 + k] * Wsh[k * 128 + n];
  fsh[t] = s;
}

// ---------------------------------------------------------------------------
// Content attention (one wave per row m in M1=65536, dl=128, L=20):
//   aw = softmax(mask(q.k/sqrt(dl))); f_caq = aw@f_w_hat * mmask;
//   f_cq = f_c_hat * (f_caq + f_s_hat)   (written over the q buffer)
// ---------------------------------------------------------------------------
__global__ __launch_bounds__(256)
void k_cattn(float* __restrict__ qc, const float* __restrict__ kc,
             const float* __restrict__ fwh, const float* __restrict__ fch,
             const float* __restrict__ fsh, const float* __restrict__ qmask,
             const float* __restrict__ mmask)
{
  int wid = (blockIdx.x << 3) + (threadIdx.x >> 5);
  if (wid >= 65536) return;
  int lane = threadIdx.x & 31;
  int m = wid, bst = m >> 2, b = m >> 12;
  size_t qoff = (size_t)m * 128 + lane * 4;
  float4 qv = *(const float4*)(qc + qoff);
  float aw[20]; float mx = -3.0e38f;
  #pragma unroll
  for (int l = 0; l < 20; l++) {
    float4 kv = *(const float4*)(kc + (size_t)(b * 20 + l) * 128 + lane * 4);
    float p = qv.x * kv.x + qv.y * kv.y + qv.z * kv.z + qv.w * kv.w;
    p = wave_sum(p) * INV_SQRT_DL;
    float qm = qmask[b * 20 + l];
    p = (qm == 0.f) ? NEGV : p * qm;
    aw[l] = p; mx = fmaxf(mx, p);
  }
  float s = 0.f;
  #pragma unroll
  for (int l = 0; l < 20; l++) { aw[l] = __expf(aw[l] - mx); s += aw[l]; }
  float rinv = 1.f / s;
  float4 fa = {0.f, 0.f, 0.f, 0.f};
  #pragma unroll
  for (int l = 0; l < 20; l++) {
    float a = aw[l] * rinv;
    float4 wv = *(const float4*)(fwh + (size_t)(b * 20 + l) * 128 + lane * 4);
    fa.x += a * wv.x; fa.y += a * wv.y; fa.z += a * wv.z; fa.w += a * wv.w;
  }
  float mmv = mmask[bst];
  float4 fc4 = *(const float4*)(fch + qoff);
  float4 fs4 = *(const float4*)(fsh + (size_t)b * 128 + lane * 4);
  float4 o;
  o.x = fc4.x * (fa.x * mmv + fs4.x);
  o.y = fc4.y * (fa.y * mmv + fs4.y);
  o.z = fc4.z * (fa.z * mmv + fs4.z);
  o.w = fc4.w * (fa.w * mmv + fs4.w);
  *(float4*)(qc + qoff) = o;
}

// ---------------------------------------------------------------------------
// Content K=4 self-attention (one wave per bst=16384):
//   A = softmax(f_cq.f_cq^T/sqrt(dl)) * mmask ; f_cc_hat = A @ f_c_hat
//   (f_cq read from qc, f_cc_hat written over qc)
// ---------------------------------------------------------------------------
__global__ __launch_bounds__(256)
void k_cAc(float* __restrict__ qc, const float* __restrict__ fch,
           const float* __restrict__ mmask)
{
  int wid = (blockIdx.x << 3) + (threadIdx.x >> 5);
  if (wid >= 16384) return;
  int lane = threadIdx.x & 31;
  int bst = wid;
  float fq[4][4];
  #pragma unroll
  for (int k = 0; k < 4; k++) {
    float4 v = *(const float4*)(qc + (size_t)(bst * 4 + k) * 128 + lane * 4);
    fq[k][0] = v.x; fq[k][1] = v.y; fq[k][2] = v.z; fq[k][3] = v.w;
  }
  float Am[4][4];
  #pragma unroll
  for (int k = 0; k < 4; k++)
    #pragma unroll
    for (int j = 0; j < 4; j++) {
      float p = fq[k][0]*fq[j][0] + fq[k][1]*fq[j][1] + fq[k][2]*fq[j][2] + fq[k][3]*fq[j][3];
      Am[k][j] = wave_sum(p) * INV_SQRT_DL;
    }
  float mmv = mmask[bst];
  #pragma unroll
  for (int k = 0; k < 4; k++) {
    float mx = fmaxf(fmaxf(Am[k][0], Am[k][1]), fmaxf(Am[k][2], Am[k][3]));
    float s = 0.f;
    #pragma unroll
    for (int j = 0; j < 4; j++) { Am[k][j] = __expf(Am[k][j] - mx); s += Am[k][j]; }
    float r = mmv / s;
    #pragma unroll
    for (int j = 0; j < 4; j++) Am[k][j] *= r;
  }
  float fh[4][4];
  #pragma unroll
  for (int j = 0; j < 4; j++) {
    float4 v = *(const float4*)(fch + (size_t)(bst * 4 + j) * 128 + lane * 4);
    fh[j][0] = v.x; fh[j][1] = v.y; fh[j][2] = v.z; fh[j][3] = v.w;
  }
  #pragma unroll
  for (int k = 0; k < 4; k++) {
    float4 o = {0.f, 0.f, 0.f, 0.f};
    #pragma unroll
    for (int j = 0; j < 4; j++) {
      float a = Am[k][j];
      o.x += a * fh[j][0]; o.y += a * fh[j][1]; o.z += a * fh[j][2]; o.w += a * fh[j][3];
    }
    *(float4*)(qc + (size_t)(bst * 4 + k) * 128 + lane * 4) = o;
  }
}

// ---------------------------------------------------------------------------
// Boundary L=20 attention (one wave per (b,t), 512 rows, D=512):
//   f_baq = softmax(mask(qb.kb/sqrt(D))) @ f_w * lmask ;
//   f_bq  = bu * (f_baq + f_s)
// ---------------------------------------------------------------------------
__global__ __launch_bounds__(256)
void k_battn(const float* __restrict__ qb, const float* __restrict__ kb,
             const float* __restrict__ fw, const float* __restrict__ bu,
             const float* __restrict__ fs, const float* __restrict__ qmask,
             const float* __restrict__ lmask, float* __restrict__ fbq)
{
  int wid = (blockIdx.x << 3) + (threadIdx.x >> 5);
  if (wid >= 512) return;
  int lane = threadIdx.x & 31;
  int b = wid >> 5;
  size_t roff = (size_t)wid * 512 + lane * 16;
  float qv[16];
  #pragma unroll
  for (int c = 0; c < 16; c++) qv[c] = qb[roff + c];
  float aw[20]; float mx = -3.0e38f;
  for (int l = 0; l < 20; l++) {
    const float* kr = kb + (size_t)(b * 20 + l) * 512 + lane * 16;
    float p = 0.f;
    #pragma unroll
    for (int c = 0; c < 16; c++) p += qv[c] * kr[c];
    p = wave_sum(p) * INV_SQRT_D;
    float qm = qmask[b * 20 + l];
    p = (qm == 0.f) ? NEGV : p * qm;
    aw[l] = p; mx = fmaxf(mx, p);
  }
  float s = 0.f;
  #pragma unroll
  for (int l = 0; l < 20; l++) { aw[l] = __expf(aw[l] - mx); s += aw[l]; }
  float rinv = 1.f / s;
  float facc[16];
  #pragma unroll
  for (int c = 0; c < 16; c++) facc[c] = 0.f;
  for (int l = 0; l < 20; l++) {
    const float* fr = fw + (size_t)(b * 20 + l) * 512 + lane * 16;
    float a = aw[l] * rinv;
    #pragma unroll
    for (int c = 0; c < 16; c++) facc[c] += a * fr[c];
  }
  float lm = lmask[wid];
  #pragma unroll
  for (int c = 0; c < 16; c++) {
    float bq  = bu[roff + c];
    float fsv = fs[(size_t)b * 512 + lane * 16 + c];
    fbq[roff + c] = bq * (facc[c] * lm + fsv);
  }
}

// ---------------------------------------------------------------------------
// Boundary T=32 self-attention + gated f_bm (one wave per (b,i)):
//   A = softmax(mask(f_bq.f_bq^T/sqrt(D))) * lmask_i
//   bu_new = (A@bu)*lmask_i + bu + sum_j A_j * sigmoid(mu*f_s)*mu
// ---------------------------------------------------------------------------
__global__ __launch_bounds__(256)
void k_bA(const float* __restrict__ fbq, const float* __restrict__ bu,
          const float* __restrict__ mu, const float* __restrict__ fs,
          const float* __restrict__ lmask, float* __restrict__ bu_tmp)
{
  int wid = (blockIdx.x << 3) + (threadIdx.x >> 5);
  if (wid >= 512) return;
  int lane = threadIdx.x & 31;
  int b = wid >> 5;
  size_t roff = (size_t)wid * 512 + lane * 16;
  float fqi[16];
  #pragma unroll
  for (int c = 0; c < 16; c++) fqi[c] = fbq[roff + c];
  float A[32]; float mx = -3.0e38f;
  for (int j = 0; j < 32; j++) {
    const float* fj = fbq + (size_t)(b * 32 + j) * 512 + lane * 16;
    float p = 0.f;
    #pragma unroll
    for (int c = 0; c < 16; c++) p += fqi[c] * fj[c];
    p = wave_sum(p) * INV_SQRT_D;
    float lmj = lmask[b * 32 + j];
    p = (lmj == 0.f) ? NEGV : p * lmj;
    A[j] = p; mx = fmaxf(mx, p);
  }
  float s = 0.f;
  #pragma unroll
  for (int j = 0; j < 32; j++) { A[j] = __expf(A[j] - mx); s += A[j]; }
  float lmi = lmask[wid];
  float scale = lmi / s;
  #pragma unroll
  for (int j = 0; j < 32; j++) A[j] *= scale;
  float fsv[16];
  #pragma unroll
  for (int c = 0; c < 16; c++) fsv[c] = fs[(size_t)b * 512 + lane * 16 + c];
  float abb[16], abm[16];
  #pragma unroll
  for (int c = 0; c < 16; c++) { abb[c] = 0.f; abm[c] = 0.f; }
  for (int j = 0; j < 32; j++) {
    float a = A[j];
    const float* br = bu + (size_t)(b * 32 + j) * 512 + lane * 16;
    const float* mr = mu + ((size_t)(wid * 32 + j)) * 512 + lane * 16;
    #pragma unroll
    for (int c = 0; c < 16; c++) {
      abb[c] += a * br[c];
      float x = mr[c];
      float g = 1.f / (1.f + __expf(-x * fsv[c]));
      abm[c] += a * g * x;
    }
  }
  #pragma unroll
  for (int c = 0; c < 16; c++)
    bu_tmp[roff + c] = abb[c] * lmi + bu[roff + c] + abm[c];
}

// ---------------------------------------------------------------------------
template<int AM, int EM>
static inline void launch_gemm(hipStream_t s, const float* A, const bf16_t* Wt,
                               const float* bias, float* C, int M, int K, int N,
                               const float* mask, const float* mu, const float* fs)
{
  int tiles  = (M / 16) * (N / 64);
  int blocks = (tiles + 7) / 8;
  k_gemm<AM, EM><<<blocks, 256, 0, s>>>(A, Wt, bias, C, M, K, N, mask, mu, fs);
}

extern "C" void kernel_launch(void* const* d_in, const int* in_sizes, int n_in,
                              void* d_out, int out_size, void* d_ws, size_t ws_size,
                              hipStream_t stream)
{
  (void)in_sizes; (void)n_in; (void)out_size; (void)ws_size;
  const float* f_c   = (const float*)d_in[0];
  const float* f_m   = (const float*)d_in[1];
  const float* f_b   = (const float*)d_in[2];
  const float* f_w   = (const float*)d_in[3];
  const float* f_s   = (const float*)d_in[4];
  const float* qmask = (const float*)d_in[5];
  const float* lmask = (const float*)d_in[6];
  const float* mmask = (const float*)d_in[7];
  const float* Wq_b  = (const float*)d_in[8];
  const float* bq_b  = (const float*)d_in[9];
  const float* Wk_b  = (const float*)d_in[10];
  const float* bk_b  = (const float*)d_in[11];
  const float* Wq_c  = (const float*)d_in[12];
  const float* bq_c  = (const float*)d_in[13];
  const float* Wk_c  = (const float*)d_in[14];
  const float* bk_c  = (const float*)d_in[15];
  const float* Wch   = (const float*)d_in[16];
  const float* bch   = (const float*)d_in[17];
  const float* Wwh   = (const float*)d_in[18];
  const float* bwh   = (const float*)d_in[19];
  const float* Wsh   = (const float*)d_in[20];
  const float* bsh   = (const float*)d_in[21];
  const float* Wcc   = (const float*)d_in[22];
  const float* bcc   = (const float*)d_in[23];
  const float* Wfb   = (const float*)d_in[24];
  const float* bfb   = (const float*)d_in[25];
  const float* Wfc   = (const float*)d_in[26];
  const float* bfc   = (const float*)d_in[27];

  float* mu = (float*)d_out;                 // (B,T,T,D) = 8388608 floats
  float* bu = (float*)d_out + 8388608;       // (B,T,D)   = 262144 floats

  char* wp = (char*)d_ws;
  auto take = [&wp](size_t bytes) -> void* {
    void* p = (void*)wp;
    wp += (bytes + 255) & ~(size_t)255;
    return p;
  };
  float*  cu    = (float*)take(134217728ull);  // content state (B,T,T,K,D)
  float*  fch   = (float*)take(33554432ull);   // f_c_hat (M1,dl)
  float*  qc    = (float*)take(33554432ull);   // q / f_cq / f_cc_hat (M1,dl)
  float*  fwh   = (float*)take(163840);        // f_w_hat (B,L,dl)
  float*  kc    = (float*)take(163840);        // k_c (B,L,dl)
  float*  fsh   = (float*)take(8192);          // f_s_hat (B,dl)
  float*  qbuf  = (float*)take(1048576);       // qb (B,T,D)
  float*  kbuf  = (float*)take(655360);        // kb (B,L,D)
  float*  fbq   = (float*)take(1048576);       // f_bq (B,T,D)
  float*  butmp = (float*)take(1048576);       // bu_new (B,T,D)
  bf16_t* WchT  = (bf16_t*)take(512 * 128 * 2);
  bf16_t* WwhT  = (bf16_t*)take(512 * 128 * 2);
  bf16_t* WqcT  = (bf16_t*)take(128 * 128 * 2);
  bf16_t* WkcT  = (bf16_t*)take(128 * 128 * 2);
  bf16_t* WccT  = (bf16_t*)take(128 * 512 * 2);
  bf16_t* WqbT  = (bf16_t*)take(512 * 512 * 2);
  bf16_t* WkbT  = (bf16_t*)take(512 * 512 * 2);
  bf16_t* WfbT  = (bf16_t*)take(512 * 512 * 2);
  bf16_t* WfcT  = (bf16_t*)take(512 * 512 * 2);

  // bf16 transposed weights (cheap; redone every launch -> deterministic)
  k_w2t<<<(512*128 + 255)/256, 256, 0, stream>>>(Wch,  WchT, 512, 128);
  k_w2t<<<(512*128 + 255)/256, 256, 0, stream>>>(Wwh,  WwhT, 512, 128);
  k_w2t<<<(128*128 + 255)/256, 256, 0, stream>>>(Wq_c, WqcT, 128, 128);
  k_w2t<<<(128*128 + 255)/256, 256, 0, stream>>>(Wk_c, WkcT, 128, 128);
  k_w2t<<<(128*512 + 255)/256, 256, 0, stream>>>(Wcc,  WccT, 128, 512);
  k_w2t<<<(512*512 + 255)/256, 256, 0, stream>>>(Wq_b, WqbT, 512, 512);
  k_w2t<<<(512*512 + 255)/256, 256, 0, stream>>>(Wk_b, WkbT, 512, 512);
  k_w2t<<<(512*512 + 255)/256, 256, 0, stream>>>(Wfb,  WfbT, 512, 512);
  k_w2t<<<(512*512 + 255)/256, 256, 0, stream>>>(Wfc,  WfcT, 512, 512);

  // state init: cu=f_c (ws), mu=f_m (d_out), bu=f_b (d_out)
  hipMemcpyAsync(cu, f_c, 134217728ull, hipMemcpyDeviceToDevice, stream);
  hipMemcpyAsync(mu, f_m, 33554432ull,  hipMemcpyDeviceToDevice, stream);
  hipMemcpyAsync(bu, f_b, 1048576ull,   hipMemcpyDeviceToDevice, stream);

  for (int it = 0; it < 3; ++it) {
    // ---------------- content unit ----------------
    launch_gemm<AM_DIRECT, EM_MASK_BST>(stream, cu,  WchT, bch, fch, 65536, 512, 128, mmask, nullptr, nullptr);
    launch_gemm<AM_DIRECT, EM_MASK_ROW>(stream, f_w, WwhT, bwh, fwh,   320, 512, 128, qmask, nullptr, nullptr);
    k_fsh<<<8, 256, 0, stream>>>(f_s, Wsh, bsh, fsh);
    launch_gemm<AM_DIRECT, EM_BIAS>(stream, fch, WqcT, bq_c, qc, 65536, 128, 128, nullptr, nullptr, nullptr);
    launch_gemm<AM_DIRECT, EM_BIAS>(stream, fwh, WkcT, bk_c, kc,   320, 128, 128, nullptr, nullptr, nullptr);
    k_cattn<<<8192, 256, 0, stream>>>(qc, kc, fwh, fch, fsh, qmask, mmask);
    k_cAc<<<2048, 256, 0, stream>>>(qc, fch, mmask);
    // cu = (f_cc_hat@Wcc+bcc)*mmask + cu + sigmoid(mu*f_s)*mu   (in place)
    launch_gemm<AM_DIRECT, EM_CC>(stream, qc, WccT, bcc, cu, 65536, 128, 512, mmask, mu, f_s);
    // ---------------- boundary unit ----------------
    launch_gemm<AM_DIRECT, EM_BIAS>(stream, bu,  WqbT, bq_b, qbuf, 512, 512, 512, nullptr, nullptr, nullptr);
    launch_gemm<AM_DIRECT, EM_BIAS>(stream, f_w, WkbT, bk_b, kbuf, 320, 512, 512, nullptr, nullptr, nullptr);
    k_battn<<<64, 256, 0, stream>>>(qbuf, kbuf, f_w, bu, f_s, qmask, lmask, fbq);
    k_bA<<<64, 256, 0, stream>>>(fbq, bu, mu, f_s, lmask, butmp);
    // ---------------- moment unit ----------------
    // mu += (outer(bu_n,bu_n)@Wfb+bfb)*mmask ; mu += (meanK(cu)@Wfc+bfc)*mmask
    launch_gemm<AM_OUTER, EM_MOM>(stream, butmp, WfbT, bfb, mu, 16384, 512, 512, mmask, nullptr, nullptr);
    launch_gemm<AM_MEAN4, EM_MOM>(stream, cu,    WfcT, bfc, mu, 16384, 512, 512, mmask, nullptr, nullptr);
    hipMemcpyAsync(bu, butmp, 1048576ull, hipMemcpyDeviceToDevice, stream);
  }
}
